// MultiHeadAttention_14723147890855
// MI455X (gfx1250) — matrix operand order repaired and checked by reading, not compiled, not run on hardware
//
#include <hip/hip_runtime.h>

// ---------------------------------------------------------------------------
// Causal multi-head attention, CDNA5 (gfx1250), wave32 WMMA f16.
//  B=2, L=4096, D=512, H=8, Dh=64. scale = 1/sqrt(512) (per reference).
//  Pipeline: qkv_proj (f32->f16 WMMA GEMM, 16x64 tiles, head-major pack)
//            attn     (flash-style, double-buffered async LDS K/V staging)
//            out_proj (f16 WMMA GEMM, 16x64 tiles -> f32 d_out)
// ---------------------------------------------------------------------------

typedef __attribute__((ext_vector_type(16))) _Float16 v16h;
typedef __attribute__((ext_vector_type(8)))  float    v8f;

#define D_MODEL 512
#define NHEADS  8
#define HDIM    64
#define LSEQ    4096
#define NBATCH  2
#define NTOK    (NBATCH * LSEQ)   // 8192 rows
#define SCALE   0.044194173824159216f  // 1/sqrt(512)

union H16 {            // bit-cast helper: v16h <-> 2x uint4 (32B, LDS b128 loads)
  v16h h;
  uint4 q[2];
};

// A-matrix (16x32 f16) K index for VGPR-pair j (0..7), lane-half (0/1).
__device__ __forceinline__ int a_kbase(int j, int half) {
  return 2 * j + 16 * half;
}

__device__ __forceinline__ v8f wmma_f16(const v16h& a, const v16h& b, const v8f& c) {
  return __builtin_amdgcn_wmma_f32_16x16x32_f16(false, a, false, b, (short)0, c,
                                                false, false);
}

// LDS byte address for DS-side operand of async-LDS ops: flat addresses of the
// LDS aperture carry the LDS offset in bits [31:0] (ISA sec 10.2), so a
// truncating cast of the generic pointer is the portable way to get it.
__device__ __forceinline__ unsigned lds_addr(const void* p) {
  return (unsigned)(unsigned long long)p;
}

// CDNA5 async copy global->LDS, 16 bytes per lane, tracked with ASYNCcnt.
__device__ __forceinline__ void async_copy_b128(const void* gaddr, unsigned ldsa) {
  asm volatile("global_load_async_to_lds_b128 %0, %1, off"
               :: "v"(ldsa), "v"(gaddr)
               : "memory");
}
__device__ __forceinline__ void wait_async0() {
  asm volatile("s_wait_asynccnt 0" ::: "memory");
}

// ---------------------------------------------------------------------------
// Kernel 1: Q/K/V projection. y = x @ W^T, f32 in, f16 out, head-major pack.
// grid = (M/16, N/64, 3), block = 32: one wave -> 16x64 tile (4 accumulators).
// ---------------------------------------------------------------------------
__global__ void __launch_bounds__(32) qkv_proj_kernel(
    const float* __restrict__ Xq, const float* __restrict__ Xk,
    const float* __restrict__ Xv, const float* __restrict__ Wq,
    const float* __restrict__ Wk, const float* __restrict__ Wv,
    _Float16* __restrict__ Qh, _Float16* __restrict__ Kh,
    _Float16* __restrict__ Vh) {
  const int z = blockIdx.z;
  const float* X = (z == 0) ? Xq : (z == 1) ? Xk : Xv;
  const float* W = (z == 0) ? Wq : (z == 1) ? Wk : Wv;
  _Float16* Out  = (z == 0) ? Qh : (z == 1) ? Kh : Vh;

  const int lane = threadIdx.x & 31;
  const int nl   = lane & 15;
  const int half = lane >> 4;
  const int tm = blockIdx.x, tn = blockIdx.y;

  const float* xrow = X + (size_t)(tm * 16 + nl) * D_MODEL;  // A: row = M
  const float* wrow[4];
#pragma unroll
  for (int cc = 0; cc < 4; ++cc)                              // B: row = N (W^T)
    wrow[cc] = W + (size_t)(tn * 64 + cc * 16 + nl) * D_MODEL;

  v8f c[4];
  const v8f vzero = {};
#pragma unroll
  for (int cc = 0; cc < 4; ++cc) c[cc] = vzero;

  for (int k0 = 0; k0 < D_MODEL; k0 += 32) {
    if (k0 + 32 < D_MODEL) __builtin_prefetch(xrow + k0 + 32, 0, 0);
    v16h a;
#pragma unroll
    for (int j = 0; j < 8; ++j) {
      const int kk = k0 + a_kbase(j, half);
      a[2 * j]     = (_Float16)xrow[kk];
      a[2 * j + 1] = (_Float16)xrow[kk + 1];
    }
#pragma unroll
    for (int cc = 0; cc < 4; ++cc) {
      v16h b;
#pragma unroll
      for (int e = 0; e < 16; ++e)
        b[e] = (_Float16)wrow[cc][k0 + 16 * half + e];
      c[cc] = wmma_f16(a, b, c[cc]);
    }
  }

#pragma unroll
  for (int cc = 0; cc < 4; ++cc)
#pragma unroll
    for (int r = 0; r < 8; ++r) {
      const int mg = tm * 16 + r + 8 * half;          // C/D row
      const int ng = tn * 64 + cc * 16 + nl;          // C/D col
      const int bb = mg >> 12, ll = mg & (LSEQ - 1);
      const int hh = ng >> 6,  dd = ng & (HDIM - 1);
      Out[(((size_t)bb * NHEADS + hh) * LSEQ + ll) * HDIM + dd] =
          (_Float16)c[cc][r];
    }
}

// ---------------------------------------------------------------------------
// Kernel 2: flash attention. grid = (L/128, B*H), block = 256 (8 waves).
// Each wave owns 16 query rows; block shares 32-row K/V tiles staged with
// double-buffered async global->LDS copies (ASYNCcnt).
// ---------------------------------------------------------------------------
__global__ void __launch_bounds__(256) attn_kernel(
    const _Float16* __restrict__ Qh, const _Float16* __restrict__ Kh,
    const _Float16* __restrict__ Vh, _Float16* __restrict__ Oh) {
  __shared__ _Float16 sK[2][32 * HDIM];     // 2 x 4 KB
  __shared__ _Float16 sV[2][32 * HDIM];     // 2 x 4 KB
  __shared__ _Float16 sP[8][16 * 32];       // 8 x 1 KB (per-wave P tiles)

  const int tid  = threadIdx.x;
  const int wv   = tid >> 5;
  const int lane = tid & 31;
  const int nl   = lane & 15;
  const int half = lane >> 4;
  const int bh   = blockIdx.y;              // b*8 + h
  const int qblock = blockIdx.x * 128;
  const int q0     = qblock + wv * 16;

  const _Float16* kbase = Kh + (size_t)bh * LSEQ * HDIM;
  const _Float16* vbase = Vh + (size_t)bh * LSEQ * HDIM;

  // Q fragments for this wave: 16x64 -> two 16x32 A-fragments.
  v16h aq[2];
  {
    const _Float16* qrow = Qh + ((size_t)bh * LSEQ + q0 + nl) * HDIM;
#pragma unroll
    for (int kc = 0; kc < 2; ++kc)
#pragma unroll
      for (int j = 0; j < 8; ++j) {
        const int d = kc * 32 + a_kbase(j, half);
        aq[kc][2 * j]     = qrow[d];
        aq[kc][2 * j + 1] = qrow[d + 1];
      }
  }

  float m_r[8], l_r[8];
  v8f acc[4];
  const v8f vzero = {};
#pragma unroll
  for (int i = 0; i < 4; ++i) acc[i] = vzero;
#pragma unroll
  for (int r = 0; r < 8; ++r) { m_r[r] = -1e30f; l_r[r] = 0.0f; }

  // Issue one 32-row K/V tile into LDS buffer `buf` (256 lanes x 16B each).
  auto issue_tile = [&](int buf, int kv0) {
    const _Float16* gk = kbase + (size_t)kv0 * HDIM + tid * 8;
    const _Float16* gv = vbase + (size_t)kv0 * HDIM + tid * 8;
    async_copy_b128(gk, lds_addr(&sK[buf][tid * 8]));
    async_copy_b128(gv, lds_addr(&sV[buf][tid * 8]));
  };

  const int kv_end = qblock + 128;          // causal bound for whole block
  issue_tile(0, 0);
  wait_async0();
  __syncthreads();

  int cur = 0;
  for (int kv0 = 0; kv0 < kv_end; kv0 += 32) {
    if (kv0 + 32 < kv_end) issue_tile(cur ^ 1, kv0 + 32);  // prefetch next

    if (kv0 <= q0 + 15) {                   // wave-uniform causal skip
      // ---- S = Q @ K^T  (16 x 32 logits, two 16x16 C-fragments) ----
      v8f s[2];
#pragma unroll
      for (int cc = 0; cc < 2; ++cc) {
        const _Float16* kr = &sK[cur][(cc * 16 + nl) * HDIM + 16 * half];
        H16 bk0, bk1;
        bk0.q[0] = *(const uint4*)(kr);
        bk0.q[1] = *(const uint4*)(kr + 8);
        bk1.q[0] = *(const uint4*)(kr + 32);
        bk1.q[1] = *(const uint4*)(kr + 40);
        v8f zz = vzero;
        zz = wmma_f16(aq[0], bk0.h, zz);
        zz = wmma_f16(aq[1], bk1.h, zz);
        s[cc] = zz;
      }

      // ---- online softmax (row reductions across 16-lane groups) ----
#pragma unroll
      for (int r = 0; r < 8; ++r) {
        const int row = q0 + r + 8 * half;
        float v0 = s[0][r] * SCALE;
        float v1 = s[1][r] * SCALE;
        if (kv0 + nl > row)      v0 = -1e30f;   // causal mask
        if (kv0 + 16 + nl > row) v1 = -1e30f;
        float rm = fmaxf(v0, v1);
#pragma unroll
        for (int m = 1; m < 16; m <<= 1)
          rm = fmaxf(rm, __shfl_xor(rm, m, 32));
        const float mnew = fmaxf(m_r[r], rm);
        const float p0 = __expf(v0 - mnew);
        const float p1 = __expf(v1 - mnew);
        float rs = p0 + p1;
#pragma unroll
        for (int m = 1; m < 16; m <<= 1)
          rs += __shfl_xor(rs, m, 32);
        const float alpha = __expf(m_r[r] - mnew);
        l_r[r] = l_r[r] * alpha + rs;
        m_r[r] = mnew;
#pragma unroll
        for (int cc = 0; cc < 4; ++cc) acc[cc][r] *= alpha;
        // stage P (C-layout -> LDS, row-major 16x32)
        sP[wv][(r + 8 * half) * 32 + nl]      = (_Float16)p0;
        sP[wv][(r + 8 * half) * 32 + 16 + nl] = (_Float16)p1;
      }
      // Wave-local LDS RAW: all DS stores must land before the gathers below.
      asm volatile("s_wait_dscnt 0" ::: "memory");

      // ---- re-gather P as A-fragment (16x32, K = 32 kv columns) ----
      v16h ap;
#pragma unroll
      for (int j = 0; j < 8; ++j) {
        const int kb = a_kbase(j, half);
        ap[2 * j]     = sP[wv][nl * 32 + kb];
        ap[2 * j + 1] = sP[wv][nl * 32 + kb + 1];
      }

      // ---- O += P @ V  (four 16x16 N-chunks over head dim) ----
#pragma unroll
      for (int cc = 0; cc < 4; ++cc) {
        v16h bv;
#pragma unroll
        for (int e = 0; e < 16; ++e)
          bv[e] = sV[cur][(e + 16 * half) * HDIM + cc * 16 + nl];
        acc[cc] = wmma_f16(ap, bv, acc[cc]);
      }
    }

    wait_async0();                          // next buffer landed in LDS
    __syncthreads();                        // everyone done with `cur`
    cur ^= 1;
  }

  // ---- normalize and store O (row-major [b*L+l][512], heads concatenated) ----
  const int bb = bh >> 3, hh = bh & 7;
#pragma unroll
  for (int cc = 0; cc < 4; ++cc)
#pragma unroll
    for (int r = 0; r < 8; ++r) {
      const int row = q0 + r + 8 * half;
      const float o = acc[cc][r] / l_r[r];
      Oh[(size_t)(bb * LSEQ + row) * D_MODEL + hh * HDIM + cc * 16 + nl] =
          (_Float16)o;
    }
}

// ---------------------------------------------------------------------------
// Kernel 3: output projection. y = O @ Wo^T, f16 in, f32 out.
// grid = (M/16, N/64), block = 32: one wave -> 16x64 tile.
// ---------------------------------------------------------------------------
__global__ void __launch_bounds__(32) out_proj_kernel(
    const _Float16* __restrict__ Oh, const float* __restrict__ Wo,
    float* __restrict__ Y) {
  const int lane = threadIdx.x & 31;
  const int nl   = lane & 15;
  const int half = lane >> 4;
  const int tm = blockIdx.x, tn = blockIdx.y;

  const _Float16* arow = Oh + (size_t)(tm * 16 + nl) * D_MODEL;
  const float* wrow[4];
#pragma unroll
  for (int cc = 0; cc < 4; ++cc)
    wrow[cc] = Wo + (size_t)(tn * 64 + cc * 16 + nl) * D_MODEL;

  v8f c[4];
  const v8f vzero = {};
#pragma unroll
  for (int cc = 0; cc < 4; ++cc) c[cc] = vzero;

  for (int k0 = 0; k0 < D_MODEL; k0 += 32) {
    if (k0 + 32 < D_MODEL) __builtin_prefetch(arow + k0 + 32, 0, 0);
    v16h a;
#pragma unroll
    for (int j = 0; j < 8; ++j) {
      const int kk = k0 + a_kbase(j, half);
      a[2 * j]     = arow[kk];
      a[2 * j + 1] = arow[kk + 1];
    }
#pragma unroll
    for (int cc = 0; cc < 4; ++cc) {
      v16h b;
#pragma unroll
      for (int e = 0; e < 16; ++e)
        b[e] = (_Float16)wrow[cc][k0 + 16 * half + e];
      c[cc] = wmma_f16(a, b, c[cc]);
    }
  }

#pragma unroll
  for (int cc = 0; cc < 4; ++cc)
#pragma unroll
    for (int r = 0; r < 8; ++r) {
      const int mg = tm * 16 + r + 8 * half;
      const int ng = tn * 64 + cc * 16 + nl;
      Y[(size_t)mg * D_MODEL + ng] = c[cc][r];
    }
}

// ---------------------------------------------------------------------------
extern "C" void kernel_launch(void* const* d_in, const int* in_sizes, int n_in,
                              void* d_out, int out_size, void* d_ws,
                              size_t ws_size, hipStream_t stream) {
  const float* q  = (const float*)d_in[0];
  const float* k  = (const float*)d_in[1];
  const float* v  = (const float*)d_in[2];
  // d_in[3] = boolean causal mask: recomputed analytically, not read.
  const float* Wq = (const float*)d_in[4];
  const float* Wk = (const float*)d_in[5];
  const float* Wv = (const float*)d_in[6];
  const float* Wo = (const float*)d_in[7];
  float* out = (float*)d_out;

  // Workspace layout (f16): Q,K,V head-major [b,h,l,64]; O row-major [m,512].
  // 4 x 8 MiB = 32 MiB total.
  _Float16* Qh = (_Float16*)d_ws;
  _Float16* Kh = Qh + (size_t)NTOK * D_MODEL;
  _Float16* Vh = Kh + (size_t)NTOK * D_MODEL;
  _Float16* Oh = Vh + (size_t)NTOK * D_MODEL;

  qkv_proj_kernel<<<dim3(NTOK / 16, D_MODEL / 64, 3), 32, 0, stream>>>(
      q, k, v, Wq, Wk, Wv, Qh, Kh, Vh);
  attn_kernel<<<dim3(LSEQ / 128, NBATCH * NHEADS), 256, 0, stream>>>(
      Qh, Kh, Vh, Oh);
  out_proj_kernel<<<dim3(NTOK / 16, D_MODEL / 64), 32, 0, stream>>>(
      Oh, Wo, out);
}
